// DCRNN_7215545057356
// MI455X (gfx1250) — matrix operand order repaired and checked by reading, not compiled, not run on hardware
//
#include <hip/hip_runtime.h>

// ---------------------------------------------------------------------------
// DCRNN on MI455X (gfx1250): all GEMMs via v_wmma_f32_16x16x32_f16.
// Activations node-major [N, B, F]: diffusion z' = L @ z is one big GEMM
// [512,512] x [512, B*F]. Dual-operand K-loop fuses the x/h split of the
// gate GEMMs; GRU pointwise math (r*h, h-update) fused into GEMM epilogues.
// Double-buffered LDS pipeline, BK=64 (16 WMMA per barrier), packed-b32
// transpose staging for B.
// ---------------------------------------------------------------------------

typedef _Float16 half_t;
typedef __attribute__((ext_vector_type(16))) _Float16       v16h;
typedef __attribute__((ext_vector_type(8)))  _Float16       v8h;
typedef __attribute__((ext_vector_type(8)))  unsigned short v8us;
typedef __attribute__((ext_vector_type(8)))  float          v8f;

#define DC_B   128
#define DC_T   24
#define DC_N   512
#define DC_F   2
#define DC_H   128
#define DC_HOR 12
#define DC_NB  (DC_N * DC_B)      // 65536 rows for gate GEMMs

#define BM 64
#define BN 128
#define BK 64
#define LDSS 72                   // BK + 8 pad (halfs): 144B row stride, conflict-free

// ---------------------------------------------------------------------------
// Dual-operand WMMA GEMM:
//   acc = A1[M,K1] @ B1[K1,N]  +  A2[M,K2] @ B2[K2,N]      (f32 accum)
//   A  : f16, element (m,k) at A[m*lda + k]
//   B  : f16, element (k,c) at B[k*ldb + (c>>bFwShift)*bG + (c & bMask)]
//   off: (m>>7)*cso + (m&127)*csi + (c>>cFwShift)*cG + (c&cMask)
//   v = act(acc + bias[c&cMask]);   act: 0 none, 1 sigmoid, 2 tanh, 3 relu
//   mode 0: C[off]=v (opt), Cb[off]=(f16)v (opt)
//   mode 1: Cb[off] = (f16)(v * P1[off])                        (rh = r*h)
//   mode 2: u=P1[off]; h=C[off]; nh=(1-u)h+u*v; C=nh; Cb=(f16)nh (GRU update)
//   fa*/fb*: per-operand uniform flags enabling branchless b128 staging
// ---------------------------------------------------------------------------
__global__ __launch_bounds__(128)
void k_gemm_f16(const half_t* __restrict__ A1, int lda1, int K1, int fa1, int fb1,
                const half_t* __restrict__ A2, int lda2, int K2, int fa2, int fb2,
                const half_t* __restrict__ B1, const half_t* __restrict__ B2,
                long long ldb, int bFwShift, long long bG,
                float* __restrict__ C, half_t* __restrict__ Cb,
                const float* __restrict__ P1,
                long long cso, long long csi, int cFwShift, long long cG,
                const float* __restrict__ bias,
                int M, int Ncols, int act, int mode)
{
    __shared__ half_t As[2][BM][LDSS];
    __shared__ half_t Bt[2][BN][LDSS];       // transposed: Bt[buf][col][k]

    const int tid   = threadIdx.x;
    const int wave  = tid >> 5;
    const int lane  = tid & 31;
    const int lidx  = lane & 15;
    const int lhalf = lane >> 4;
    const int wm    = wave >> 1;             // 2x2 wave grid; wave tile 32 x 64
    const int wn    = wave & 1;
    const int m0    = blockIdx.x * BM;
    const int n0    = blockIdx.y * BN;
    const unsigned bMask = (1u << bFwShift) - 1u;
    const unsigned cMask = (1u << cFwShift) - 1u;

    const int t1 = (K1 + BK - 1) / BK;
    const int t2 = (K2 + BK - 1) / BK;       // K2 may be 0
    const int nt = t1 + t2;

    v8f accv[2][4];
#pragma unroll
    for (int i = 0; i < 2; ++i)
#pragma unroll
        for (int j = 0; j < 4; ++j) accv[i][j] = (v8f)0.0f;

    auto stage = [&](int ti, int buf) {
        const bool ph2 = (ti >= t1);
        const half_t* Ap = ph2 ? A2 : A1;
        const half_t* Bp = ph2 ? B2 : B1;
        const int lda = ph2 ? lda2 : lda1;
        const int Kp  = ph2 ? K2  : K1;
        const int fa  = ph2 ? fa2 : fa1;
        const int fb  = ph2 ? fb2 : fb1;
        const int k0  = (ph2 ? ti - t1 : ti) * BK;

        // ---- A tile: 64 rows x 64 k ----
        if (fa) {                            // 512 b128 / 128 thr, branchless
#pragma unroll
            for (int i = 0; i < 4; ++i) {
                int idx = tid + i * 128;
                int r   = idx >> 3;
                int kk  = (idx & 7) << 3;
                *(v8h*)&As[buf][r][kk] =
                    *(const v8h*)(Ap + (size_t)(m0 + r) * lda + (k0 + kk));
            }
            int kp = k0 + BK;
            if (kp < Kp)                     // global_prefetch_b8 for next tile
                __builtin_prefetch(Ap + (size_t)(m0 + (tid >> 1)) * lda + kp, 0, 1);
        } else {                             // u32-or-zero (K, lda always even)
            for (int i = 0; i < 16; ++i) {
                int idx = tid + i * 128;
                int r   = idx >> 5;
                int kk  = (idx & 31) << 1;
                int gm  = m0 + r, gk = k0 + kk;
                unsigned v = 0u;
                if (gm < M && gk + 2 <= Kp)
                    v = *(const unsigned*)(Ap + (size_t)gm * lda + gk);
                *(unsigned*)&As[buf][r][kk] = v;
            }
        }

        // ---- B tile: 64 k x 128 cols, stored transposed ----
        if (fb) {                            // 2 k-rows/thread/iter, b32 packed
#pragma unroll
            for (int i = 0; i < 4; ++i) {
                int idx = tid + i * 128;
                int kk  = (idx >> 4) << 1;   // 0,2,...,62
                int cc  = (idx & 15) << 3;
                int gc  = n0 + cc;
                size_t off = (size_t)(k0 + kk) * ldb +
                             (size_t)((unsigned)gc >> bFwShift) * bG +
                             (size_t)((unsigned)gc & bMask);
                v8us va = *(const v8us*)(Bp + off);
                v8us vb = *(const v8us*)(Bp + off + ldb);
#pragma unroll
                for (int j = 0; j < 8; ++j) {
                    unsigned pv = (unsigned)va[j] | ((unsigned)vb[j] << 16);
                    *(unsigned*)&Bt[buf][cc + j][kk] = pv;   // (k,k+1) pair
                }
            }
        } else {
            for (int i = 0; i < 32; ++i) {
                int idx = tid + i * 128;
                int kk  = idx >> 6;
                int cc  = (idx & 63) << 1;
                int gk  = k0 + kk, gc = n0 + cc;
                unsigned v = 0u;
                if (gk < Kp && gc + 2 <= Ncols) {
                    size_t off = (size_t)gk * ldb +
                                 (size_t)((unsigned)gc >> bFwShift) * bG +
                                 (size_t)((unsigned)gc & bMask);
                    v = *(const unsigned*)(Bp + off);
                }
                *(unsigned short*)&Bt[buf][cc][kk]     = (unsigned short)(v & 0xffffu);
                *(unsigned short*)&Bt[buf][cc + 1][kk] = (unsigned short)(v >> 16);
            }
        }
    };

    stage(0, 0);
    __syncthreads();

    int buf = 0;
    for (int ti = 0; ti < nt; ++ti) {
        if (ti + 1 < nt) stage(ti + 1, buf ^ 1);   // pipeline next tile

#pragma unroll
        for (int kh = 0; kh < BK; kh += 32) {      // two 16x16x32 K-chunks
            v16h av[2];
#pragma unroll
            for (int sm = 0; sm < 2; ++sm) {
                int row = wm * 32 + sm * 16 + lidx;
                int ko  = kh + lhalf * 8;
                v8h a0 = *(const v8h*)&As[buf][row][ko];
                v8h a1 = *(const v8h*)&As[buf][row][ko + 16];
                av[sm] = __builtin_shufflevector(a0, a1, 0,1,2,3,4,5,6,7,8,9,10,11,12,13,14,15);
            }
            v16h bv[4];
#pragma unroll
            for (int sn = 0; sn < 4; ++sn) {
                int col = wn * 64 + sn * 16 + lidx;
                int kb  = kh + lhalf * 8;
                v8h b0 = *(const v8h*)&Bt[buf][col][kb];
                v8h b1 = *(const v8h*)&Bt[buf][col][kb + 16];
                bv[sn] = __builtin_shufflevector(b0, b1, 0,1,2,3,4,5,6,7,8,9,10,11,12,13,14,15);
            }
#pragma unroll
            for (int sm = 0; sm < 2; ++sm)
#pragma unroll
                for (int sn = 0; sn < 4; ++sn)
                    accv[sm][sn] = __builtin_amdgcn_wmma_f32_16x16x32_f16(
                        false, av[sm], false, bv[sn], (short)0, accv[sm][sn], false, false);
        }
        __syncthreads();
        buf ^= 1;
    }

    // ---- epilogue ----
#pragma unroll
    for (int sm = 0; sm < 2; ++sm) {
#pragma unroll
        for (int sn = 0; sn < 4; ++sn) {
#pragma unroll
            for (int r = 0; r < 8; ++r) {
                int gm = m0 + wm * 32 + sm * 16 + lhalf * 8 + r;
                int gc = n0 + wn * 64 + sn * 16 + lidx;
                if (gm < M && gc < Ncols) {
                    float v = accv[sm][sn][r];
                    size_t off = (size_t)((unsigned)gm >> 7) * cso +
                                 (size_t)((unsigned)gm & 127u) * csi +
                                 (size_t)((unsigned)gc >> cFwShift) * cG +
                                 (size_t)((unsigned)gc & cMask);
                    if (bias) v += bias[(unsigned)gc & cMask];
                    if (act == 1)      v = 1.0f / (1.0f + __expf(-v));
                    else if (act == 2) v = tanhf(v);
                    else if (act == 3) v = v > 0.0f ? v : 0.0f;
                    if (mode == 0) {
                        if (C)  C[off]  = v;
                        if (Cb) Cb[off] = (half_t)v;
                    } else if (mode == 1) {            // rh = sigmoid(.) * h
                        Cb[off] = (half_t)(v * P1[off]);
                    } else {                           // GRU: h=(1-u)h+u*c
                        float u  = P1[off];
                        float h  = C[off];
                        float nh = (1.0f - u) * h + u * v;
                        C[off]  = nh;
                        Cb[off] = (half_t)nh;
                    }
                }
            }
        }
    }
}

// ---------------------------------------------------------------------------
// Elementwise helpers
// ---------------------------------------------------------------------------
__global__ void k_cast_f32_f16(const float* __restrict__ s, half_t* __restrict__ d, int n) {
    int i = blockIdx.x * blockDim.x + threadIdx.x;
    if (i < n) d[i] = (half_t)s[i];
}

// x [B,T,N,F] f32  ->  xall [T,N,B,F] f16
__global__ void k_prep_x(const float* __restrict__ x, half_t* __restrict__ xall, int total) {
    int i = blockIdx.x * blockDim.x + threadIdx.x;
    if (i >= total) return;
    int f = i % DC_F;       int tmp = i / DC_F;
    int b = tmp % DC_B;     tmp /= DC_B;
    int n = tmp % DC_N;     int t = tmp / DC_N;
    size_t src = (((size_t)b * DC_T + t) * DC_N + n) * DC_F + f;
    xall[i] = (half_t)x[src];
}

__global__ void k_zero2(float* __restrict__ hf, half_t* __restrict__ hh, int n) {
    int i = blockIdx.x * blockDim.x + threadIdx.x;
    if (i < n) { hf[i] = 0.0f; hh[i] = (half_t)0.0f; }
}

// copy [rows, Fin] contiguous f16 into segment 0 of [rows, Fd] buffer
__global__ void k_copyseg(const half_t* __restrict__ src, half_t* __restrict__ dst,
                          int rows, int Fin, int Fd) {
    int i = blockIdx.x * blockDim.x + threadIdx.x;
    if (i >= rows * Fin) return;
    int r = i / Fin, f = i % Fin;
    dst[(size_t)r * Fd + f] = src[i];
}

// ---------------------------------------------------------------------------
// Host side
// ---------------------------------------------------------------------------
static inline void gemm2(hipStream_t s,
                         const half_t* A1, int lda1, int K1,
                         const half_t* A2, int lda2, int K2,
                         const half_t* B1, const half_t* B2,
                         long long ldb, int bFwShift, long long bG,
                         float* C, half_t* Cb, const float* P1,
                         long long cso, long long csi, int cFwShift, long long cG,
                         const float* bias, int M, int Ncols, int act, int mode)
{
    int colsOK = (Ncols % BN == 0) && (bFwShift >= 7) && (ldb % 8 == 0) && (bG % 8 == 0);
    int fa1 = (M % BM == 0) && (K1 % BK == 0) && (lda1 % 8 == 0);
    int fb1 = (K1 % BK == 0) && colsOK;
    int fa2 = (K2 > 0) && (M % BM == 0) && (K2 % BK == 0) && (lda2 % 8 == 0);
    int fb2 = (K2 > 0) && (K2 % BK == 0) && colsOK;
    dim3 grid((M + BM - 1) / BM, (Ncols + BN - 1) / BN);
    k_gemm_f16<<<grid, 128, 0, s>>>(A1, lda1, K1, fa1, fb1,
                                    A2, lda2, K2, fa2, fb2,
                                    B1, B2, ldb, bFwShift, bG,
                                    C, Cb, P1, cso, csi, cFwShift, cG,
                                    bias, M, Ncols, act, mode);
}

static inline dim3 g1(int n) { return dim3((n + 255) / 256); }

extern "C" void kernel_launch(void* const* d_in, const int* in_sizes, int n_in,
                              void* d_out, int out_size, void* d_ws, size_t ws_size,
                              hipStream_t stream)
{
    (void)out_size; (void)ws_size;
    const float* x = (const float*)d_in[0];
    const float* L = (const float*)d_in[1];

    // --- resolve pytree flattening order (insertion vs sorted keys) ---
    int iW[2][3], iB[2][3];   // [layer][Wr,Wu,Wc] / [br,bu,bc]
    for (int l = 0; l < 2; ++l) {
        int base = 2 + 6 * l;
        bool sortedP = (n_in >= base + 4) && (in_sizes[base + 1] == in_sizes[base]);
        if (sortedP) { // Wc, Wr, Wu, bc, br, bu
            iW[l][0] = base + 1; iW[l][1] = base + 2; iW[l][2] = base + 0;
            iB[l][0] = base + 4; iB[l][1] = base + 5; iB[l][2] = base + 3;
        } else {       // Wr, br, Wu, bu, Wc, bc
            iW[l][0] = base + 0; iW[l][1] = base + 2; iW[l][2] = base + 4;
            iB[l][0] = base + 1; iB[l][1] = base + 3; iB[l][2] = base + 5;
        }
    }
    int db = 14;
    bool sortedD = (n_in >= 16) && (in_sizes[db + 1] != 128);
    int iW1 = db + 0, iW2, ib1, ib2;
    if (sortedD) { iW2 = db + 1; ib1 = db + 2; ib2 = db + 3; }  // W1,W2,b1,b2
    else         { ib1 = db + 1; iW2 = db + 2; ib2 = db + 3; }  // W1,b1,W2,b2

    // --- carve workspace ---
    char* p = (char*)d_ws;
    auto alloc = [&](size_t bytes) -> void* {
        void* r = (void*)p; p += (bytes + 255) & ~(size_t)255; return r;
    };
    half_t* L16    = (half_t*)alloc((size_t)DC_N * DC_N * 2);
    half_t* xall   = (half_t*)alloc((size_t)DC_T * DC_NB * DC_F * 2);
    half_t* W16[2][3]; const float* bf[2][3];
    for (int l = 0; l < 2; ++l) {
        int ind = (l == 0 ? 3 * DC_F : 3 * DC_H) + 3 * DC_H;   // 390 / 768
        for (int g = 0; g < 3; ++g) {
            W16[l][g] = (half_t*)alloc((size_t)ind * DC_H * 2);
            bf[l][g]  = (const float*)d_in[iB[l][g]];
        }
    }
    half_t* W1_16 = (half_t*)alloc((size_t)DC_H * DC_H * 2);
    half_t* W2_16 = (half_t*)alloc((size_t)DC_H * DC_HOR * 2);
    float*  hF[2]; half_t* hH[2];
    for (int l = 0; l < 2; ++l) {
        hF[l] = (float*)alloc((size_t)DC_NB * DC_H * 4);
        hH[l] = (half_t*)alloc((size_t)DC_NB * DC_H * 2);
    }
    half_t* xdiff = (half_t*)alloc((size_t)DC_NB * (3 * DC_H) * 2);  // max Fd=384
    half_t* hdiff = (half_t*)alloc((size_t)DC_NB * (3 * DC_H) * 2);
    float*  ubuf  = (float*)alloc((size_t)DC_NB * DC_H * 4);
    half_t* rh16  = (half_t*)alloc((size_t)DC_NB * DC_H * 2);
    half_t* hid16 = (half_t*)alloc((size_t)DC_NB * DC_H * 2);

    // --- setup: casts / transposes / state init ---
    k_cast_f32_f16<<<g1(DC_N * DC_N), 256, 0, stream>>>(L, L16, DC_N * DC_N);
    k_prep_x<<<g1(DC_T * DC_NB * DC_F), 256, 0, stream>>>(x, xall, DC_T * DC_NB * DC_F);
    for (int l = 0; l < 2; ++l) {
        int ind = (l == 0 ? 3 * DC_F : 3 * DC_H) + 3 * DC_H;
        for (int g = 0; g < 3; ++g)
            k_cast_f32_f16<<<g1(ind * DC_H), 256, 0, stream>>>(
                (const float*)d_in[iW[l][g]], W16[l][g], ind * DC_H);
        k_zero2<<<g1(DC_NB * DC_H), 256, 0, stream>>>(hF[l], hH[l], DC_NB * DC_H);
    }
    k_cast_f32_f16<<<g1(DC_H * DC_H), 256, 0, stream>>>((const float*)d_in[iW1], W1_16, DC_H * DC_H);
    k_cast_f32_f16<<<g1(DC_H * DC_HOR), 256, 0, stream>>>((const float*)d_in[iW2], W2_16, DC_H * DC_HOR);

    const long long sH = DC_H;
    // diffusion: buf[seg+1] = L @ buf[seg], columns grouped (Fw=Fin, G=Fd)
    auto diffuse2 = [&](half_t* buf, int Fin, int FwShift) {
        int Fd = 3 * Fin;
        long long rowS = (long long)DC_B * Fd;
        for (int s = 0; s < 2; ++s) {
            gemm2(stream, L16, DC_N, DC_N, nullptr, 0, 0,
                  buf + s * Fin, nullptr, rowS, FwShift, Fd,
                  nullptr, buf + (s + 1) * Fin, nullptr,
                  128LL * rowS, rowS, FwShift, Fd,
                  nullptr, DC_N, DC_B * Fin, 0, 0);
        }
    };

    // --- time loop ---
    for (int t = 0; t < DC_T; ++t) {
        for (int l = 0; l < 2; ++l) {
            const int Fin  = (l == 0) ? DC_F : DC_H;
            const int FwSh = (l == 0) ? 1 : 7;
            const int Fdx  = 3 * Fin;
            const half_t* inp = (l == 0) ? (xall + (size_t)t * DC_NB * DC_F) : hH[0];

            // x_diff = [inp, L inp, L^2 inp]
            k_copyseg<<<g1(DC_NB * Fin), 256, 0, stream>>>(inp, xdiff, DC_NB, Fin, Fdx);
            diffuse2(xdiff, Fin, FwSh);
            // h_diff
            k_copyseg<<<g1(DC_NB * DC_H), 256, 0, stream>>>(hH[l], hdiff, DC_NB, DC_H, 3 * DC_H);
            diffuse2(hdiff, DC_H, 7);

            // r gate (dual-K, sigmoid, fused rh = r*h -> f16)
            gemm2(stream, xdiff, Fdx, Fdx, hdiff, 3 * DC_H, 3 * DC_H,
                  W16[l][0], W16[l][0] + (size_t)Fdx * DC_H, sH, 20, 0,
                  nullptr, rh16, hF[l],
                  128LL * sH, sH, 20, 0,
                  bf[l][0], DC_NB, DC_H, 1 /*sigmoid*/, 1 /*rh*/);
            // u gate (dual-K, sigmoid -> f32)
            gemm2(stream, xdiff, Fdx, Fdx, hdiff, 3 * DC_H, 3 * DC_H,
                  W16[l][1], W16[l][1] + (size_t)Fdx * DC_H, sH, 20, 0,
                  ubuf, nullptr, nullptr,
                  128LL * sH, sH, 20, 0,
                  bf[l][1], DC_NB, DC_H, 1 /*sigmoid*/, 0);

            // rh_diff (reuse hdiff)
            k_copyseg<<<g1(DC_NB * DC_H), 256, 0, stream>>>(rh16, hdiff, DC_NB, DC_H, 3 * DC_H);
            diffuse2(hdiff, DC_H, 7);

            // c gate (dual-K, tanh) fused with GRU update: h=(1-u)h+u*c
            gemm2(stream, xdiff, Fdx, Fdx, hdiff, 3 * DC_H, 3 * DC_H,
                  W16[l][2], W16[l][2] + (size_t)Fdx * DC_H, sH, 20, 0,
                  hF[l], hH[l], ubuf,
                  128LL * sH, sH, 20, 0,
                  bf[l][2], DC_NB, DC_H, 2 /*tanh*/, 2 /*GRU*/);
        }
    }

    // --- decoder: hid = relu(h1 @ W1 + b1); out = hid @ W2 + b2 ([B,N,HOR]) ---
    gemm2(stream, hH[1], DC_H, DC_H, nullptr, 0, 0,
          W1_16, nullptr, sH, 20, 0,
          nullptr, hid16, nullptr,
          128LL * sH, sH, 20, 0,
          (const float*)d_in[ib1], DC_NB, DC_H, 3 /*relu*/, 0);
    gemm2(stream, hid16, DC_H, DC_H, nullptr, 0, 0,
          W2_16, nullptr, DC_HOR, 20, 0,
          (float*)d_out, nullptr, nullptr,
          (long long)DC_HOR, (long long)DC_N * DC_HOR, 20, 0,
          (const float*)d_in[ib2], DC_NB, DC_HOR, 0, 0);
}
